// GBSTokenizer_44461501449124
// MI455X (gfx1250) — matrix operand order repaired and checked by reading, not compiled, not run on hardware
//
#include <hip/hip_runtime.h>
#include <hip/hip_bf16.h>
#include <stdint.h>

// ---------------------------------------------------------------------------
// GBS tokenizer for MI455X (gfx1250, wave32, WMMA).
// Conv-as-GEMM (16320x2304x768) + projection GEMM (16320x768x768) on
// v_wmma_f32_16x16x32_bf16, fp32 accumulate. Weights pre-transposed to
// fragment-major [o][k] so all LDS staging is b128. Conv stages a 66-row halo
// A-tile + 3-tap B-tile for K=64 per barrier pair -> 12 WMMA/wave/stage.
// Routing collapses analytically (softmax over size-1 axis == 1.0 exactly)
// to one scalar blend constant.
// ---------------------------------------------------------------------------

typedef __bf16 bf16_t;
typedef bf16_t v16bf __attribute__((ext_vector_type(16)));
typedef float  v8f   __attribute__((ext_vector_type(8)));

#define DD     768
#define LL     2040
#define NBATCH 8
#define KCONV  (3 * DD)

union FragBF { uint4 q[2]; v16bf v; };

__device__ __forceinline__ unsigned short f2bf(float f) {
  union { float f; unsigned int u; } x; x.f = f;
  unsigned int r = x.u + 0x7FFFu + ((x.u >> 16) & 1u);   // round-to-nearest-even
  return (unsigned short)(r >> 16);
}

// ---------------- converts / repacks (all one-time, tiny) ----------------

__global__ void __launch_bounds__(256) cvt_x_bf16(const float* __restrict__ X,
                                                  unsigned short* __restrict__ Xb,
                                                  int n) {
  int i = blockIdx.x * 256 + threadIdx.x;
  if (i < n) Xb[i] = f2bf(X[i]);
}

// conv_w (O=768, I=768, T=3) -> W2T[o][t*768 + i]  (fragment-major, bf16)
__global__ void __launch_bounds__(256) cvt_wconv_T(const float* __restrict__ W,
                                                   unsigned short* __restrict__ W2T) {
  int i = blockIdx.x * 256 + threadIdx.x;
  if (i >= DD * KCONV) return;
  int o  = i / KCONV;
  int r  = i % KCONV;
  int t  = r / DD;
  int ci = r % DD;
  W2T[i] = f2bf(W[(o * DD + ci) * 3 + t]);
}

// wd (k,o) row-major -> WdT[o][k] (fragment-major, bf16)
__global__ void __launch_bounds__(256) cvt_wd_T(const float* __restrict__ W,
                                                unsigned short* __restrict__ WdT) {
  int i = blockIdx.x * 256 + threadIdx.x;
  if (i >= DD * DD) return;
  int o = i / DD;
  int k = i % DD;
  WdT[i] = f2bf(W[k * DD + o]);
}

// ---------------- conv-as-GEMM (K = 3 taps x 768) ----------------
// Block: 256 threads = 8 waves (4x2), tile 64 rows(L) x 64 cols(Dout).
// Per K=64 chunk: 66-row halo A stage + 3-tap B stage, then each wave issues
// 12 WMMA (2 K-steps x 3 taps x 2 col-strips) with A-fragment reuse.

__global__ void __launch_bounds__(256) conv_gemm(const unsigned short* __restrict__ Xb,
                                                 const unsigned short* __restrict__ W2T,
                                                 const float* __restrict__ bias,
                                                 float* __restrict__ Y) {
  __shared__ __align__(16) unsigned short sA[66 * 72];    // halo rows, stride 144B
  __shared__ __align__(16) unsigned short sB[64 * 200];   // [o][tap*64 + k], stride 400B

  const int tid  = threadIdx.x;
  const int wid  = tid >> 5;        // wave32
  const int lane = tid & 31;
  const int hs   = lane >> 4;       // half-wave select
  const int ml   = lane & 15;
  const int wr   = wid >> 1;        // 0..3
  const int wc   = wid & 1;         // 0..1

  const int o0 = blockIdx.x * 64;
  const int l0 = blockIdx.y * 64;
  const int nz = blockIdx.z;

  v8f c0 = {0.f,0.f,0.f,0.f,0.f,0.f,0.f,0.f};
  v8f c1 = {0.f,0.f,0.f,0.f,0.f,0.f,0.f,0.f};

  // static staging assignments
  const int arow0 = tid >> 3, achk0 = tid & 7;            // unit tid
  const int arow1 = (tid + 256) >> 3, achk1 = tid & 7;    // unit tid+256
  const int arow2 = 64 + (tid >> 3);                      // units 512..527 (tid<16)

  for (int kb = 0; kb < DD; kb += 64) {
    __syncthreads();
    // ---- stage A halo: 66 rows x 8 chunks of 8 halfs = 528 b128 units ----
    {
      const int lsrc = l0 + arow0 - 1;
      uint4 av = make_uint4(0u, 0u, 0u, 0u);
      if (lsrc >= 0 && lsrc < LL) {
        const unsigned short* p = Xb + ((size_t)nz * LL + lsrc) * DD + kb + achk0 * 8;
        av = *(const uint4*)p;
        if (kb + 64 < DD) __builtin_prefetch(p + 64, 0, 1);
      }
      *(uint4*)&sA[arow0 * 72 + achk0 * 8] = av;
    }
    {
      const int lsrc = l0 + arow1 - 1;
      uint4 av = make_uint4(0u, 0u, 0u, 0u);
      if (lsrc >= 0 && lsrc < LL)
        av = *(const uint4*)(Xb + ((size_t)nz * LL + lsrc) * DD + kb + achk1 * 8);
      *(uint4*)&sA[arow1 * 72 + achk1 * 8] = av;
    }
    if (tid < 16) {
      const int lsrc = l0 + arow2 - 1;
      uint4 av = make_uint4(0u, 0u, 0u, 0u);
      if (lsrc >= 0 && lsrc < LL)
        av = *(const uint4*)(Xb + ((size_t)nz * LL + lsrc) * DD + kb + achk0 * 8);
      *(uint4*)&sA[arow2 * 72 + achk0 * 8] = av;
    }
    // ---- stage B: 3 taps x 64 o x 8 chunks = 1536 b128 units (6/thread) ----
    #pragma unroll
    for (int j = 0; j < 6; ++j) {
      const int u = tid + 256 * j;
      const int tap = u >> 9, rem = u & 511, o = rem >> 3, chk = rem & 7;
      uint4 bv = *(const uint4*)(W2T + (size_t)(o0 + o) * KCONV + tap * DD + kb + chk * 8);
      *(uint4*)&sB[o * 200 + tap * 64 + chk * 8] = bv;
    }
    __syncthreads();

    #pragma unroll
    for (int ks = 0; ks < 64; ks += 32) {
      #pragma unroll
      for (int tap = 0; tap < 3; ++tap) {
        FragBF a, b0, b1;
        // A fragment: sA row (m + tap) == global row l0 + m + tap - 1
        const unsigned short* ap = &sA[(wr * 16 + ml + tap) * 72 + ks + hs * 8];
        a.q[0] = *(const uint4*)(ap);
        a.q[1] = *(const uint4*)(ap + 16);
        const unsigned short* bp = &sB[(wc * 32 + ml) * 200 + tap * 64 + ks + hs * 8];
        b0.q[0] = *(const uint4*)(bp);
        b0.q[1] = *(const uint4*)(bp + 16);
        const unsigned short* bq = bp + 16 * 200;
        b1.q[0] = *(const uint4*)(bq);
        b1.q[1] = *(const uint4*)(bq + 16);
        c0 = __builtin_amdgcn_wmma_f32_16x16x32_bf16(false, a.v, false, b0.v, (short)0, c0, false, false);
        c1 = __builtin_amdgcn_wmma_f32_16x16x32_bf16(false, a.v, false, b1.v, (short)0, c1, false, false);
      }
    }
  }

  // epilogue: C layout -> lanes 0-15: M=r, lanes 16-31: M=r+8 ; N = ml
  const int col0 = o0 + wc * 32 + ml;
  const int col1 = col0 + 16;
  const float bb0 = bias[col0];
  const float bb1 = bias[col1];
  #pragma unroll
  for (int r = 0; r < 8; ++r) {
    const int m = r + 8 * hs;
    const int l = l0 + wr * 16 + m;
    if (l < LL) {
      const size_t row = ((size_t)nz * LL + l) * DD;
      Y[row + col0] = c0[r] + bb0;
      Y[row + col1] = c1[r] + bb1;
    }
  }
}

// ---------------- multi-scale pool + blend ----------------
// X_hat[n,l,:] = c * ( x[n,l,:] + m2[l/2] + m3[l/3] + m4[l/4] ), groups of 12.

__global__ void __launch_bounds__(256) pool_blend(const float* __restrict__ Xc,
                                                  unsigned short* __restrict__ Xh,
                                                  float c) {
  const int g  = blockIdx.x;            // 0 .. N*(L/12)-1
  const int n  = g / (LL / 12);
  const int lg = g % (LL / 12);
  const size_t base = ((size_t)n * LL + (size_t)lg * 12) * DD;

  for (int d = threadIdx.x; d < DD; d += 256) {
    float x[12];
    #pragma unroll
    for (int j = 0; j < 12; ++j) x[j] = Xc[base + (size_t)j * DD + d];
    float m2[6], m3[4], m4[3];
    #pragma unroll
    for (int p = 0; p < 6; ++p) m2[p] = (x[2*p] + x[2*p+1]) / 2.0f;
    #pragma unroll
    for (int p = 0; p < 4; ++p) m3[p] = (x[3*p] + x[3*p+1] + x[3*p+2]) / 3.0f;
    #pragma unroll
    for (int p = 0; p < 3; ++p) m4[p] = (x[4*p] + x[4*p+1] + x[4*p+2] + x[4*p+3]) / 4.0f;
    #pragma unroll
    for (int j = 0; j < 12; ++j) {
      float v = c * (x[j] + m2[j >> 1] + m3[j / 3] + m4[j >> 2]);
      Xh[base + (size_t)j * DD + d] = f2bf(v);
    }
  }
}

// ---------------- output projection GEMM (16320 x 768 x 768) ----------------
// Same wave layout; stages K=64 per barrier pair -> 4 WMMA per wave per stage.

__global__ void __launch_bounds__(256) out_gemm(const unsigned short* __restrict__ Ab,
                                                const unsigned short* __restrict__ WdT,
                                                const float* __restrict__ bias,
                                                float* __restrict__ Y) {
  __shared__ __align__(16) unsigned short sA[64 * 72];   // 64 rows x K64, stride 144B
  __shared__ __align__(16) unsigned short sB[64 * 72];   // 64 o    x K64, stride 144B

  const int tid  = threadIdx.x;
  const int wid  = tid >> 5;
  const int lane = tid & 31;
  const int hs   = lane >> 4;
  const int ml   = lane & 15;
  const int wr   = wid >> 1;
  const int wc   = wid & 1;

  const int o0 = blockIdx.x * 64;
  const int r0 = blockIdx.y * 64;   // flat row over N*L (255*64 == 16320 exactly)

  v8f c0 = {0.f,0.f,0.f,0.f,0.f,0.f,0.f,0.f};
  v8f c1 = {0.f,0.f,0.f,0.f,0.f,0.f,0.f,0.f};

  for (int kb = 0; kb < DD; kb += 64) {
    __syncthreads();
    // stage A: 64 rows x 8 chunks = 512 b128 units (2 per thread)
    #pragma unroll
    for (int j = 0; j < 2; ++j) {
      const int u = tid + 256 * j;
      const int row = u >> 3, chk = u & 7;
      const unsigned short* p = Ab + (size_t)(r0 + row) * DD + kb + chk * 8;
      uint4 av = *(const uint4*)p;
      if (kb + 64 < DD) __builtin_prefetch(p + 64, 0, 1);
      *(uint4*)&sA[row * 72 + chk * 8] = av;
    }
    // stage B: 64 o x 8 chunks = 512 b128 units (already transposed)
    #pragma unroll
    for (int j = 0; j < 2; ++j) {
      const int u = tid + 256 * j;
      const int row = u >> 3, chk = u & 7;
      uint4 bv = *(const uint4*)(WdT + (size_t)(o0 + row) * DD + kb + chk * 8);
      *(uint4*)&sB[row * 72 + chk * 8] = bv;
    }
    __syncthreads();

    #pragma unroll
    for (int ks = 0; ks < 64; ks += 32) {
      FragBF a, b0, b1;
      const unsigned short* ap = &sA[(wr * 16 + ml) * 72 + ks + hs * 8];
      a.q[0] = *(const uint4*)(ap);
      a.q[1] = *(const uint4*)(ap + 16);
      const unsigned short* bp = &sB[(wc * 32 + ml) * 72 + ks + hs * 8];
      b0.q[0] = *(const uint4*)(bp);
      b0.q[1] = *(const uint4*)(bp + 16);
      const unsigned short* bq = bp + 16 * 72;
      b1.q[0] = *(const uint4*)(bq);
      b1.q[1] = *(const uint4*)(bq + 16);
      c0 = __builtin_amdgcn_wmma_f32_16x16x32_bf16(false, a.v, false, b0.v, (short)0, c0, false, false);
      c1 = __builtin_amdgcn_wmma_f32_16x16x32_bf16(false, a.v, false, b1.v, (short)0, c1, false, false);
    }
  }

  const int col0 = o0 + wc * 32 + ml;
  const int col1 = col0 + 16;
  const float bb0 = bias[col0];
  const float bb1 = bias[col1];
  #pragma unroll
  for (int r = 0; r < 8; ++r) {
    const int m = r + 8 * hs;
    const size_t row = (size_t)(r0 + wr * 16 + m) * DD;
    Y[row + col0] = c0[r] + bb0;
    Y[row + col1] = c1[r] + bb1;
  }
}

// ---------------- launch ----------------

extern "C" void kernel_launch(void* const* d_in, const int* in_sizes, int n_in,
                              void* d_out, int out_size, void* d_ws, size_t ws_size,
                              hipStream_t stream) {
  (void)in_sizes; (void)n_in; (void)out_size; (void)ws_size;
  const float* X      = (const float*)d_in[0];
  const float* conv_w = (const float*)d_in[1];
  const float* conv_b = (const float*)d_in[2];
  // d_in[3] = wr, d_in[4] = br : mathematically dead (softmax over size-1 axis == 1.0)
  const float* wd     = (const float*)d_in[5];
  const float* bd     = (const float*)d_in[6];
  float* out = (float*)d_out;

  char* ws = (char*)d_ws;
  size_t off = 0;
  auto take = [&](size_t bytes) -> char* {
    char* p = ws + off;
    off = (off + bytes + 255) & ~(size_t)255;
    return p;
  };
  const size_t NX = (size_t)NBATCH * LL * DD;           // 12,533,760
  unsigned short* Xbf  = (unsigned short*)take(NX * 2);                  // X in bf16
  unsigned short* W2T  = (unsigned short*)take((size_t)DD * KCONV * 2);  // conv W, [o][k]
  unsigned short* WdT  = (unsigned short*)take((size_t)DD * DD * 2);     // wd, [o][k]
  float*          Xc   = (float*)take(NX * 4);                           // conv out, fp32
  unsigned short* Xhat = (unsigned short*)take(NX * 2);                  // blended, bf16

  cvt_x_bf16<<<(int)((NX + 255) / 256), 256, 0, stream>>>(X, Xbf, (int)NX);
  cvt_wconv_T<<<(DD * KCONV + 255) / 256, 256, 0, stream>>>(conv_w, W2T);
  cvt_wd_T<<<(DD * DD + 255) / 256, 256, 0, stream>>>(wd, WdT);

  dim3 g1(DD / 64, (LL + 63) / 64, NBATCH);   // (12, 32, 8)
  conv_gemm<<<g1, 256, 0, stream>>>(Xbf, W2T, conv_b, Xc);

  // Calibrated routing weight: P[n,l,m] = sum_{k<L} fl(1/L) * 1.0 (fp32, sequential)
  float c = 0.0f;
  for (int k = 0; k < LL; ++k) c += 1.0f / (float)LL;

  pool_blend<<<NBATCH * (LL / 12), 256, 0, stream>>>(Xc, Xhat, c);

  dim3 g2(DD / 64, (NBATCH * LL) / 64, 1);    // (12, 255)
  out_gemm<<<g2, 256, 0, stream>>>(Xhat, WdT, bd, out);
}
